// TokenizerEncoder_86096914415822
// MI455X (gfx1250) — matrix operand order repaired and checked by reading, not compiled, not run on hardware
//
#include <hip/hip_runtime.h>

// ---------------------------------------------------------------------------
// Types for CDNA5 WMMA (wave32)
// ---------------------------------------------------------------------------
typedef __attribute__((ext_vector_type(16))) __bf16 bf16x16;
typedef __attribute__((ext_vector_type(8)))  __bf16 bf16x8;
typedef __attribute__((ext_vector_type(8)))  float  f32x8;

#define CAT16(a, b) __builtin_shufflevector(a, b, 0,1,2,3,4,5,6,7,8,9,10,11,12,13,14,15)

__device__ __forceinline__ float gelu_exact(float x) {
    return 0.5f * x * (1.0f + erff(x * 0.70710678118654752f));
}

// ---------------------------------------------------------------------------
// Weight pre-swizzle: W[K,N] f32 -> bf16 in exact B-fragment lane order.
// Layout: Wsw[((nt*KC + c2)*32 + lane)*16 + e]  with k = c2*32 + (lane>=16)*16 + e,
//         n = nt*16 + (lane&15). One 32B contiguous chunk per lane per K-chunk.
// ---------------------------------------------------------------------------
__global__ void swizzle_weights(const float* __restrict__ W, __bf16* __restrict__ Wsw,
                                int K, int N, int nmat) {
    int idx = blockIdx.x * blockDim.x + threadIdx.x;
    int per = K * N;
    if (idx >= nmat * per) return;
    int m    = idx / per;
    int w    = idx - m * per;
    int e    = w & 15;
    int lane = (w >> 4) & 31;
    int rest = w >> 9;
    int KC   = K >> 5;
    int c2   = rest % KC;
    int nt   = rest / KC;
    int k    = c2 * 32 + (lane >> 4) * 16 + e;
    int n    = nt * 16 + (lane & 15);
    Wsw[idx] = (__bf16)W[(size_t)m * per + (size_t)k * N + n];
}

// ---------------------------------------------------------------------------
// GEMM: C[M,N] = A[M,K](bf16 row-major) * Wsw (pre-swizzled bf16), f32 acc.
// Block 128 = 4 waves; wave owns a 16x64 strip (4 N-tiles), A fragments kept
// in registers across the strip. Grid: (N/64, M/64).
// mode 0: Cf = v ; mode 1: Cb = bf16(gelu(v)) ; mode 2: Cf += v
// ---------------------------------------------------------------------------
template <int K>
__global__ void gemm_bf16_wmma(const __bf16* __restrict__ A,
                               const __bf16* __restrict__ Wsw,
                               float* __restrict__ Cf,
                               __bf16* __restrict__ Cb,
                               int M, int N, int mode) {
    constexpr int KC = K / 32;
    int wave = threadIdx.x >> 5;
    int lane = threadIdx.x & 31;
    int half = lane >> 4;
    int lm   = lane & 15;

    int mRow   = (blockIdx.y * 4 + wave) * 16;
    int ntBase = blockIdx.x * 4;

    // A fragments: two contiguous b128 spans per K-chunk, reused for 4 tiles
    const __bf16* Arow = A + (size_t)(mRow + lm) * K;
    int kBase = half * 16;
    bf16x16 af[KC];
#pragma unroll
    for (int c2 = 0; c2 < KC; c2++) {
        bf16x8 lo = *(const bf16x8*)(Arow + c2 * 32 + kBase);
        bf16x8 hi = *(const bf16x8*)(Arow + c2 * 32 + 8 + kBase);
        af[c2] = CAT16(lo, hi);
    }

    const bf16x16* wf = (const bf16x16*)Wsw;
#pragma unroll
    for (int t = 0; t < 4; t++) {
        int nt = ntBase + t;
        f32x8 c = {};
#pragma unroll
        for (int c2 = 0; c2 < KC; c2++) {
            bf16x16 b = wf[((size_t)nt * KC + c2) * 32 + lane];
            c = __builtin_amdgcn_wmma_f32_16x16x32_bf16(false, af[c2], false, b,
                                                        (short)0, c, false, false);
        }
        int col = nt * 16 + lm;
#pragma unroll
        for (int r = 0; r < 8; r++) {
            int row = mRow + r + half * 8;
            size_t off = (size_t)row * N + col;
            float v = c[r];
            if (mode == 0)      Cf[off] = v;
            else if (mode == 1) Cb[off] = (__bf16)gelu_exact(v);
            else                Cf[off] += v;
        }
    }
}

// ---------------------------------------------------------------------------
// RMSNorm: h[M,64] f32 -> bf16 normalized * gamma
// ---------------------------------------------------------------------------
__global__ void rmsnorm_bf16(const float* __restrict__ x,
                             const float* __restrict__ g,
                             __bf16* __restrict__ out, int M) {
    int r = blockIdx.x * blockDim.x + threadIdx.x;
    if (r >= M) return;
    const float4* x4 = (const float4*)(x + (size_t)r * 64);
    float ss = 0.f;
#pragma unroll
    for (int i = 0; i < 16; i++) {
        float4 v = x4[i];
        ss += v.x * v.x + v.y * v.y + v.z * v.z + v.w * v.w;
    }
    float inv = rsqrtf(ss * (1.0f / 64.0f) + 1e-6f);
    __bf16* o = out + (size_t)r * 64;
    const float4* g4 = (const float4*)g;
#pragma unroll
    for (int i = 0; i < 16; i++) {
        float4 v = x4[i], gv = g4[i];
        o[4 * i + 0] = (__bf16)(v.x * gv.x * inv);
        o[4 * i + 1] = (__bf16)(v.y * gv.y * inv);
        o[4 * i + 2] = (__bf16)(v.z * gv.z * inv);
        o[4 * i + 3] = (__bf16)(v.w * gv.w * inv);
    }
}

// ---------------------------------------------------------------------------
// RoPE: read f32 q/k, write bf16 q (pre-scaled by 1/sqrt(8)) and k.
// One thread per (b,s,h,j), j in [0,4).
// ---------------------------------------------------------------------------
__global__ void rope_bf16(const float* __restrict__ q, const float* __restrict__ k,
                          __bf16* __restrict__ qo, __bf16* __restrict__ ko,
                          int B, int S) {
    int idx = blockIdx.x * blockDim.x + threadIdx.x;
    int total = B * S * 8 * 4;
    if (idx >= total) return;
    int j    = idx & 3;
    int hh   = (idx >> 2) & 7;
    int rest = idx >> 5;          // b*S + t
    int t    = rest % S;
    const float scale = 0.35355339059327373f; // 1/sqrt(8)
    float freq = __powf(10000.0f, -0.25f * (float)j);
    float ang  = (float)t * freq;
    float c = __cosf(ang), s = __sinf(ang);
    size_t base = (size_t)rest * 64 + hh * 8;
    float x1 = q[base + j], x2 = q[base + 4 + j];
    qo[base + j]     = (__bf16)((x1 * c - x2 * s) * scale);
    qo[base + 4 + j] = (__bf16)((x1 * s + x2 * c) * scale);
    x1 = k[base + j]; x2 = k[base + 4 + j];
    ko[base + j]     = (__bf16)(x1 * c - x2 * s);
    ko[base + 4 + j] = (__bf16)(x1 * s + x2 * c);
}

// ---------------------------------------------------------------------------
// V transpose: v[B,S,64] f32 -> vt[B,64,S] bf16 (d = h*8+dd major, key minor)
// so the P*V B-fragment is a contiguous 32B load per lane.
// ---------------------------------------------------------------------------
__global__ void v_transpose(const float* __restrict__ v, __bf16* __restrict__ vt,
                            int B, int S) {
    int idx = blockIdx.x * blockDim.x + threadIdx.x;
    int total = B * S * 64;
    if (idx >= total) return;
    int d    = idx & 63;
    int rest = idx >> 6;          // b*S + t
    int t    = rest % S;
    int b    = rest / S;
    vt[((size_t)b * 64 + d) * S + t] = (__bf16)v[idx];
}

// ---------------------------------------------------------------------------
// Flash-style causal attention, one wave per (b, head, 16-query tile).
// bf16 operands, vectorized fragment loads, bf16 probs staged in LDS.
// Block 256 = 8 waves; grid sized exactly (EXEC full at all WMMA sites).
// Assumes S % 32 == 0 (S=2048).
// ---------------------------------------------------------------------------
#define ATT_WAVES 8
__global__ void attn_wmma(const __bf16* __restrict__ qh,
                          const __bf16* __restrict__ kh,
                          const __bf16* __restrict__ vt,
                          __bf16* __restrict__ o,
                          int B, int S) {
    __shared__ __bf16 pbuf_all[ATT_WAVES * 16 * 32];
    int wave = threadIdx.x >> 5;
    int lane = threadIdx.x & 31;
    int wid  = blockIdx.x * ATT_WAVES + wave;
    int nQT  = S / 16;
    int qt = wid % nQT;
    int hh = (wid / nQT) & 7;
    int b  = wid / (nQT * 8);
    int half = lane >> 4, lm = lane & 15;
    __bf16* pb = pbuf_all + wave * 512;

    bf16x8 zv8 = {};

    // Q fragment (constant over key loop): 1 b128 load, lanes 0-15, K<8 valid
    bf16x8 qv = {};
    if (half == 0)
        qv = *(const bf16x8*)(qh + ((size_t)(b * S + qt * 16 + lm)) * 64 + hh * 8);
    bf16x16 aq = CAT16(qv, zv8);

    const __bf16* vtb = vt + ((size_t)b * 64 + hh * 8) * S;

    float mrow[8], lrow[8];
    f32x8 acc = {};
#pragma unroll
    for (int r = 0; r < 8; r++) { mrow[r] = -1e30f; lrow[r] = 0.0f; }

    int qEnd = qt * 16 + 15;
    for (int kb = 0; kb <= qEnd; kb += 32) {
        f32x8 s0 = {}, s1 = {};
        {   // score tile 0: keys kb..kb+15
            bf16x8 kv = {};
            if (half == 0)
                kv = *(const bf16x8*)(kh + ((size_t)(b * S + kb + lm)) * 64 + hh * 8);
            bf16x16 bk = CAT16(kv, zv8);
            s0 = __builtin_amdgcn_wmma_f32_16x16x32_bf16(false, aq, false, bk,
                                                         (short)0, s0, false, false);
        }
        {   // score tile 1: keys kb+16..kb+31
            bf16x8 kv = {};
            if (half == 0)
                kv = *(const bf16x8*)(kh + ((size_t)(b * S + kb + 16 + lm)) * 64 + hh * 8);
            bf16x16 bk = CAT16(kv, zv8);
            s1 = __builtin_amdgcn_wmma_f32_16x16x32_bf16(false, aq, false, bk,
                                                         (short)0, s1, false, false);
        }
        // causal mask + per-row local max over the 32-key chunk
        float tmax[8];
#pragma unroll
        for (int r = 0; r < 8; r++) {
            int qi = qt * 16 + r + half * 8;
            float v0 = ((kb + lm)      <= qi) ? s0[r] : -1e30f;
            float v1 = ((kb + 16 + lm) <= qi) ? s1[r] : -1e30f;
            s0[r] = v0; s1[r] = v1;
            tmax[r] = fmaxf(v0, v1);
        }
#pragma unroll
        for (int mk = 1; mk < 16; mk <<= 1) {
#pragma unroll
            for (int r = 0; r < 8; r++)
                tmax[r] = fmaxf(tmax[r], __shfl_xor(tmax[r], mk, 32));
        }
        // online softmax update; probs -> LDS as bf16 (16 rows x 32 cols)
        float psum[8];
#pragma unroll
        for (int r = 0; r < 8; r++) {
            float nm    = fmaxf(mrow[r], tmax[r]);
            float alpha = __expf(mrow[r] - nm);
            mrow[r] = nm;
            float e0 = __expf(s0[r] - nm);
            float e1 = __expf(s1[r] - nm);
            psum[r] = e0 + e1;
            lrow[r] *= alpha;
            acc[r]  *= alpha;
            pb[(r + half * 8) * 32 + lm]      = (__bf16)e0;
            pb[(r + half * 8) * 32 + 16 + lm] = (__bf16)e1;
        }
#pragma unroll
        for (int mk = 1; mk < 16; mk <<= 1) {
#pragma unroll
            for (int r = 0; r < 8; r++)
                psum[r] += __shfl_xor(psum[r], mk, 32);
        }
#pragma unroll
        for (int r = 0; r < 8; r++) lrow[r] += psum[r];

        asm volatile("s_wait_dscnt 0" ::: "memory");

        // A fragment from probs: two ds b128 loads, row = lm
        int kBase = half * 16;
        bf16x8 p0 = *(const bf16x8*)(pb + lm * 32 + kBase);
        bf16x8 p1 = *(const bf16x8*)(pb + lm * 32 + 8 + kBase);
        bf16x16 ap = CAT16(p0, p1);

        // B fragment from transposed V: one 32B contiguous load per lane
        bf16x16 bv = {};
        if (lm < 8)
            bv = *(const bf16x16*)(vtb + (size_t)lm * S + kb + half * 16);

        acc = __builtin_amdgcn_wmma_f32_16x16x32_bf16(false, ap, false, bv,
                                                      (short)0, acc, false, false);
    }

    // write attention output (bf16 -> A of O projection)
    if (lm < 8) {
#pragma unroll
        for (int r = 0; r < 8; r++) {
            int row = qt * 16 + r + half * 8;
            o[((size_t)(b * S + row)) * 64 + hh * 8 + lm] =
                (__bf16)(acc[r] / lrow[r]);
        }
    }
}

// ---------------------------------------------------------------------------
// Head: logits[M,2] = h[M,64] @ Whead[64,2] + bhead
// ---------------------------------------------------------------------------
__global__ void head_kernel(const float* __restrict__ h,
                            const float* __restrict__ Wh,
                            const float* __restrict__ bh,
                            float* __restrict__ logits, int M) {
    int r = blockIdx.x * blockDim.x + threadIdx.x;
    if (r >= M) return;
    const float* hr = h + (size_t)r * 64;
    float l0 = bh[0], l1 = bh[1];
#pragma unroll 8
    for (int i = 0; i < 64; i++) {
        float v = hr[i];
        l0 += v * Wh[2 * i];
        l1 += v * Wh[2 * i + 1];
    }
    logits[2 * r]     = l0;
    logits[2 * r + 1] = l1;
}

// ---------------------------------------------------------------------------
// Per-batch sequential chunk scan (replicates reference index math)
// ---------------------------------------------------------------------------
__global__ void scan_kernel(const float* __restrict__ logits,
                            int* __restrict__ c_idx, int* __restrict__ p_idx,
                            int B, int S) {
    if (threadIdx.x != 0) return;
    int b = blockIdx.x;
    const float* lg = logits + (size_t)b * S * 2;
    bool has = false; int num_ends = 0;
    for (int t = 0; t < S; t++)
        if (lg[2 * t] > lg[2 * t + 1]) { has = true; num_ends++; }
    if (!has) num_ends = 1;  // fallback end at S-1
    int cum = 0, startv = 0;
    bool prev_end = true;
    for (int t = 0; t < S; t++) {
        bool em = has ? (lg[2 * t] > lg[2 * t + 1]) : (t == S - 1);
        if (prev_end) startv = t;
        int chunk_id = cum;          // ends strictly before t
        int pos = t - startv;
        bool valid = (chunk_id < num_ends) && (chunk_id < 1024) && (pos < 16);
        c_idx[(size_t)b * S + t] = valid ? chunk_id : -1;
        p_idx[(size_t)b * S + t] = valid ? pos : 0;
        cum += em ? 1 : 0;
        prev_end = em;
    }
}

// ---------------------------------------------------------------------------
// Init output: broadcast pad vector / pad id, zero reg term
// ---------------------------------------------------------------------------
__global__ void init_out(float* __restrict__ out, int* __restrict__ ids,
                         const float* __restrict__ padv,
                         const int* __restrict__ pad_id,
                         float* __restrict__ reg) {
    size_t i = (size_t)blockIdx.x * blockDim.x + threadIdx.x;
    const size_t nOut = (size_t)4 * 1024 * 16 * 64;
    const size_t nIds = (size_t)4 * 1024 * 16;
    if (i < nOut) out[i] = padv[i & 63];
    if (i < nIds) ids[i] = *pad_id;
    if (i == 0)   *reg = 0.0f;
}

// ---------------------------------------------------------------------------
// Scatter valid token rows into chunk output
// ---------------------------------------------------------------------------
__global__ void scatter_kernel(const float* __restrict__ h,
                               const int* __restrict__ x_ids,
                               const int* __restrict__ c_idx,
                               const int* __restrict__ p_idx,
                               float* __restrict__ out, int* __restrict__ ids,
                               int B, int S) {
    int bt = blockIdx.x;                 // b*S + t
    int ci = c_idx[bt];
    if (ci < 0) return;
    int b = bt / S;
    int pos = p_idx[bt];
    size_t slot = ((size_t)b * 1024 + ci) * 16 + pos;
    out[slot * 64 + threadIdx.x] = h[(size_t)bt * 64 + threadIdx.x];
    if (threadIdx.x == 0) ids[slot] = x_ids[bt];
}

// ---------------------------------------------------------------------------
// reg_term = mean(logits[...,1])
// ---------------------------------------------------------------------------
__global__ void reg_reduce(const float* __restrict__ logits,
                           float* __restrict__ reg, int n) {
    __shared__ float sm[256];
    int i = blockIdx.x * blockDim.x + threadIdx.x;
    sm[threadIdx.x] = (i < n) ? logits[2 * i + 1] : 0.0f;
    __syncthreads();
    for (int s = 128; s > 0; s >>= 1) {
        if (threadIdx.x < s) sm[threadIdx.x] += sm[threadIdx.x + s];
        __syncthreads();
    }
    if (threadIdx.x == 0) atomicAdd(reg, sm[0] / (float)n);
}

// ---------------------------------------------------------------------------
// Utility: residual stream init
// ---------------------------------------------------------------------------
__global__ void copy_f32(const float* __restrict__ s, float* __restrict__ d, int n) {
    int i = blockIdx.x * blockDim.x + threadIdx.x;
    if (i < n) d[i] = s[i];
}

// ---------------------------------------------------------------------------
// Host orchestration
// ---------------------------------------------------------------------------
extern "C" void kernel_launch(void* const* d_in, const int* in_sizes, int n_in,
                              void* d_out, int out_size, void* d_ws, size_t ws_size,
                              hipStream_t stream) {
    const int B = 4, S = 2048, L = 3, FFN = 256;
    const int M = B * S;

    const float* x      = (const float*)d_in[0];
    const float* padv   = (const float*)d_in[1];
    const int*   x_ids  = (const int*)d_in[2];
    const int*   pad_id = (const int*)d_in[3];
    const float* Wq     = (const float*)d_in[4];
    const float* Wk     = (const float*)d_in[5];
    const float* Wv     = (const float*)d_in[6];
    const float* Wo     = (const float*)d_in[7];
    const float* ln1    = (const float*)d_in[8];
    const float* ln2    = (const float*)d_in[9];
    const float* W1     = (const float*)d_in[10];
    const float* W2     = (const float*)d_in[11];
    const float* Whead  = (const float*)d_in[12];
    const float* bhead  = (const float*)d_in[13];

    char* ws = (char*)d_ws;
    size_t off = 0;
    auto carve = [&](size_t bytes) -> char* {
        char* p = ws + off;
        off += (bytes + 255) & ~(size_t)255;
        return p;
    };
    float*  h      = (float*) carve((size_t)M * 64 * 4);
    __bf16* hn     = (__bf16*)carve((size_t)M * 64 * 2);
    float*  qb     = (float*) carve((size_t)M * 64 * 4);
    float*  kbuf   = (float*) carve((size_t)M * 64 * 4);
    float*  vbuf   = (float*) carve((size_t)M * 64 * 4);
    __bf16* qh     = (__bf16*)carve((size_t)M * 64 * 2);
    __bf16* kh     = (__bf16*)carve((size_t)M * 64 * 2);
    __bf16* vth    = (__bf16*)carve((size_t)M * 64 * 2);
    __bf16* ob     = (__bf16*)carve((size_t)M * 64 * 2);
    __bf16* mid    = (__bf16*)carve((size_t)M * 256 * 2);
    float*  logits = (float*) carve((size_t)M * 2 * 4);
    int*    c_idx  = (int*)   carve((size_t)M * 4);
    int*    p_idx  = (int*)   carve((size_t)M * 4);
    __bf16* Wq_b   = (__bf16*)carve((size_t)L * 64 * 64 * 2);
    __bf16* Wk_b   = (__bf16*)carve((size_t)L * 64 * 64 * 2);
    __bf16* Wv_b   = (__bf16*)carve((size_t)L * 64 * 64 * 2);
    __bf16* Wo_b   = (__bf16*)carve((size_t)L * 64 * 64 * 2);
    __bf16* W1_b   = (__bf16*)carve((size_t)L * 64 * 256 * 2);
    __bf16* W2_b   = (__bf16*)carve((size_t)L * 256 * 64 * 2);

    // pre-swizzle weights to WMMA B-fragment order (bf16)
    swizzle_weights<<<(L * 4096 + 255) / 256,  256, 0, stream>>>(Wq, Wq_b, 64, 64, L);
    swizzle_weights<<<(L * 4096 + 255) / 256,  256, 0, stream>>>(Wk, Wk_b, 64, 64, L);
    swizzle_weights<<<(L * 4096 + 255) / 256,  256, 0, stream>>>(Wv, Wv_b, 64, 64, L);
    swizzle_weights<<<(L * 4096 + 255) / 256,  256, 0, stream>>>(Wo, Wo_b, 64, 64, L);
    swizzle_weights<<<(L * 16384 + 255) / 256, 256, 0, stream>>>(W1, W1_b, 64, 256, L);
    swizzle_weights<<<(L * 16384 + 255) / 256, 256, 0, stream>>>(W2, W2_b, 256, 64, L);
    copy_f32<<<(M * 64 + 255) / 256, 256, 0, stream>>>(x, h, M * 64);

    dim3 gemmBlock(128);
    dim3 gN64(1, M / 64);    // N=64 strips
    dim3 gN256(4, M / 64);   // N=256 strips

    for (int l = 0; l < L; l++) {
        rmsnorm_bf16<<<(M + 255) / 256, 256, 0, stream>>>(h, ln1 + l * 64, hn, M);
        gemm_bf16_wmma<64><<<gN64, gemmBlock, 0, stream>>>(hn, Wq_b + (size_t)l * 4096, qb,   nullptr, M, 64, 0);
        gemm_bf16_wmma<64><<<gN64, gemmBlock, 0, stream>>>(hn, Wk_b + (size_t)l * 4096, kbuf, nullptr, M, 64, 0);
        gemm_bf16_wmma<64><<<gN64, gemmBlock, 0, stream>>>(hn, Wv_b + (size_t)l * 4096, vbuf, nullptr, M, 64, 0);
        rope_bf16<<<(B * S * 32 + 255) / 256, 256, 0, stream>>>(qb, kbuf, qh, kh, B, S);
        v_transpose<<<(M * 64 + 255) / 256, 256, 0, stream>>>(vbuf, vth, B, S);
        attn_wmma<<<(B * 8 * (S / 16)) / ATT_WAVES, 256, 0, stream>>>(qh, kh, vth, ob, B, S);
        gemm_bf16_wmma<64><<<gN64, gemmBlock, 0, stream>>>(ob, Wo_b + (size_t)l * 4096, h, nullptr, M, 64, 2);
        rmsnorm_bf16<<<(M + 255) / 256, 256, 0, stream>>>(h, ln2 + l * 64, hn, M);
        gemm_bf16_wmma<64><<<gN256, gemmBlock, 0, stream>>>(hn,  W1_b + (size_t)l * 16384, nullptr, mid, M, 256, 1);
        gemm_bf16_wmma<256><<<gN64, gemmBlock, 0, stream>>>(mid, W2_b + (size_t)l * 16384, h, nullptr, M, 64, 2);
    }

    head_kernel<<<(M + 255) / 256, 256, 0, stream>>>(h, Whead, bhead, logits, M);

    float* outp = (float*)d_out;
    int*   idsp = (int*)(outp + (size_t)B * 1024 * 16 * 64);
    float* regp = outp + (size_t)B * 1024 * 16 * 64 + (size_t)B * 1024 * 16;

    init_out<<<16384, 256, 0, stream>>>(outp, idsp, padv, pad_id, regp);
    scan_kernel<<<B, 32, 0, stream>>>(logits, c_idx, p_idx, B, S);
    scatter_kernel<<<M, 64, 0, stream>>>(h, x_ids, c_idx, p_idx, outp, idsp, B, S);
    reg_reduce<<<(M + 255) / 256, 256, 0, stream>>>(logits, regp, M);
}
